// MPN_32212254720524
// MI455X (gfx1250) — matrix operand order repaired and checked by reading, not compiled, not run on hardware
//
#include <hip/hip_runtime.h>
#include <stdint.h>

// ---------------- CDNA5 / gfx1250 MPN (chemprop D-MPNN) ----------------
// MI455X reasoning: workload is gather-bandwidth bound (~6 random 448-wide
// message rows per bond per iteration). message kept in bf16 (halves HBM/L2
// gather traffic, feeds v_wmma_f32_16x16x32_bf16 directly). Weights packed
// once into WMMA B-fragment order (W_h packed = 400KB -> L2 resident).
// GEMMs: 4-wide N-tile blocking so one LDS A-fragment feeds 4 WMMAs.
// Buffers row-padded to tile multiples -> unconditional epilogue stores.

#define HID       448
#define NTILES    (HID / 16)   // 28
#define ATOM_DIM  38
#define BOND_FDIM 49           // ATOM_DIM + BOND_DIM
#define MAX_NB    6
#define WAVES     8            // 8 wave32 per block
#define N_ITERS   4            // DEPTH - 1

typedef __attribute__((ext_vector_type(16))) __bf16 v16bf;
typedef __attribute__((ext_vector_type(8)))  float  v8f;

union Frag {
    uint4 q[2];
    v16bf v;
    unsigned short s[16];
};

__device__ __forceinline__ unsigned short f32_to_bf16_bits(float f) {
    unsigned u = __float_as_uint(f);
    unsigned r = u + 0x7FFFu + ((u >> 16) & 1u);   // round-to-nearest-even
    return (unsigned short)(r >> 16);
}
__device__ __forceinline__ unsigned pack_bf16x2(float lo, float hi) {
    return (unsigned)f32_to_bf16_bits(lo) | ((unsigned)f32_to_bf16_bits(hi) << 16);
}
__device__ __forceinline__ float bf16lo_f32(unsigned u) { return __uint_as_float(u << 16); }
__device__ __forceinline__ float bf16hi_f32(unsigned u) { return __uint_as_float(u & 0xFFFF0000u); }

__device__ __forceinline__ v8f wmma_bf16(const Frag& a, const Frag& b, v8f c) {
    return __builtin_amdgcn_wmma_f32_16x16x32_bf16(
        /*neg_a=*/false, a.v, /*neg_b=*/false, b.v,
        /*c_mod=*/(short)0, c, /*reuse_a=*/false, /*reuse_b=*/false);
}

__device__ __forceinline__ void load_bfrag(Frag& b, const unsigned short* packed,
                                           int kt, int nt, int lane) {
    const uint4* bp = (const uint4*)(packed + ((size_t)(kt * NTILES + nt) * 512) + lane * 16);
    b.q[0] = bp[0];
    b.q[1] = bp[1];
}

// ---------------------------------------------------------------------------
// Pack f32 row-major weight [K_src][448] into WMMA B-fragment order (bf16).
// B 32x16 bf16: lane<16 -> N=lane,   elem j = K(kt*32 + (j<8 ? j : 8+j))
//               lane>=16 -> N=lane-16, elem j = K(kt*32 + (j<8 ? 8+j : 16+j))
// ---------------------------------------------------------------------------
__global__ void pack_weight_kernel(const float* __restrict__ W,
                                   unsigned short* __restrict__ dst,
                                   int K_src, int kt_count) {
    int idx = blockIdx.x * blockDim.x + threadIdx.x;
    int total = kt_count * NTILES * 512;
    if (idx >= total) return;
    int j    = idx & 15;
    int lane = (idx >> 4) & 31;
    int frag = idx >> 9;
    int nt   = frag % NTILES;
    int kt   = frag / NTILES;
    int n    = nt * 16 + (lane & 15);
    int k    = kt * 32 + ((lane < 16) ? 0 : 8) + ((j < 8) ? j : (8 + j));
    float v  = (k < K_src) ? W[(size_t)k * HID + n] : 0.0f;
    dst[idx] = f32_to_bf16_bits(v);
}

// ---------------------------------------------------------------------------
// binput = fbonds @ W_i ; message0 = relu(binput) (bf16)
// Buffers padded to tile multiple -> no store guards.
// ---------------------------------------------------------------------------
__global__ __launch_bounds__(WAVES * 32)
void binput_kernel(const float* __restrict__ fbonds,
                   const unsigned short* __restrict__ Wi_p,
                   float* __restrict__ binput,
                   unsigned short* __restrict__ msg,
                   int n_bonds, int n_tiles) {
    int lane = threadIdx.x & 31;
    int wave = threadIdx.x >> 5;
    int tile = blockIdx.x * WAVES + wave;
    if (tile >= n_tiles) return;
    int m0 = tile * 16;

    int arow = min(m0 + (lane & 15), n_bonds - 1);
    const float* src = fbonds + (size_t)arow * BOND_FDIM;

    Frag a[2];
    #pragma unroll
    for (int kt = 0; kt < 2; ++kt) {
        int k0 = kt * 32 + ((lane < 16) ? 0 : 8);
        #pragma unroll
        for (int j = 0; j < 8; ++j) {
            int ka = k0 + j, kb = k0 + 16 + j;
            a[kt].s[j]     = (ka < BOND_FDIM) ? f32_to_bf16_bits(src[ka]) : (unsigned short)0;
            a[kt].s[8 + j] = (kb < BOND_FDIM) ? f32_to_bf16_bits(src[kb]) : (unsigned short)0;
        }
    }

    int rb = (lane < 16) ? 0 : 8;
    #pragma unroll 1
    for (int ntc = 0; ntc < NTILES; ntc += 4) {
        v8f acc[4];
        #pragma unroll
        for (int q = 0; q < 4; ++q) acc[q] = (v8f){0.f,0.f,0.f,0.f,0.f,0.f,0.f,0.f};
        #pragma unroll
        for (int kt = 0; kt < 2; ++kt) {
            #pragma unroll
            for (int q = 0; q < 4; ++q) {
                Frag b;
                load_bfrag(b, Wi_p, kt, ntc + q, lane);
                acc[q] = wmma_bf16(a[kt], b, acc[q]);
            }
        }
        #pragma unroll
        for (int q = 0; q < 4; ++q) {
            int col = (ntc + q) * 16 + (lane & 15);
            #pragma unroll
            for (int i = 0; i < 8; ++i) {
                size_t off = (size_t)(m0 + rb + i) * HID + col;
                float v = acc[q][i];
                binput[off] = v;
                msg[off]    = f32_to_bf16_bits(fmaxf(v, 0.f));
            }
        }
    }
}

// ---------------------------------------------------------------------------
// message_out = relu(binput + gather_sum(message_in, bgraph) @ W_h)
// Gather-sum (f32) -> LDS bf16 [16][448] per wave -> WMMA (14 K-steps,
// one A fragment reused by 4 WMMAs).
// ---------------------------------------------------------------------------
__global__ __launch_bounds__(WAVES * 32)
void mpn_iter_kernel(const unsigned short* __restrict__ msg_in,
                     const float* __restrict__ binput,
                     const unsigned short* __restrict__ Wh_p,
                     const long long* __restrict__ bgraph,
                     unsigned short* __restrict__ msg_out,
                     int n_bonds, int n_tiles) {
    __shared__ unsigned short s_nei[WAVES][16][HID];   // 114688 B of 320KB WGP LDS
    int lane = threadIdx.x & 31;
    int wave = threadIdx.x >> 5;
    int tile = blockIdx.x * WAVES + wave;
    if (tile >= n_tiles) return;
    int m0 = tile * 16;

    // ---- gather-sum neighbors into LDS (bf16 pairs) ----
    for (int r = 0; r < 16; ++r) {
        int row = min(m0 + r, n_bonds - 1);
        const long long* bg = bgraph + (size_t)row * MAX_NB;
        float a0[7], a1[7];
        #pragma unroll
        for (int c = 0; c < 7; ++c) { a0[c] = 0.f; a1[c] = 0.f; }
        #pragma unroll
        for (int k = 0; k < MAX_NB; ++k) {
            size_t nb = (size_t)bg[k];
            const unsigned* sp = (const unsigned*)(msg_in + nb * HID);
            #pragma unroll
            for (int c = 0; c < 7; ++c) {
                unsigned u = sp[c * 32 + lane];
                a0[c] += bf16lo_f32(u);
                a1[c] += bf16hi_f32(u);
            }
        }
        unsigned* drow = (unsigned*)&s_nei[wave][r][0];
        #pragma unroll
        for (int c = 0; c < 7; ++c) drow[c * 32 + lane] = pack_bf16x2(a0[c], a1[c]);
    }
    asm volatile("s_wait_dscnt 0" ::: "memory");

    // ---- GEMM: nei @ W_h, add binput, relu, store bf16 ----
    int rb = (lane < 16) ? 0 : 8;
    const unsigned short* arow_base = &s_nei[wave][lane & 15][(lane < 16) ? 0 : 8];
    #pragma unroll 1
    for (int ntc = 0; ntc < NTILES; ntc += 4) {
        v8f acc[4];
        #pragma unroll
        for (int q = 0; q < 4; ++q) acc[q] = (v8f){0.f,0.f,0.f,0.f,0.f,0.f,0.f,0.f};
        #pragma unroll 2
        for (int kt = 0; kt < HID / 32; ++kt) {
            Frag a;
            a.q[0] = *(const uint4*)(arow_base + kt * 32);
            a.q[1] = *(const uint4*)(arow_base + kt * 32 + 16);
            #pragma unroll
            for (int q = 0; q < 4; ++q) {
                Frag b;
                load_bfrag(b, Wh_p, kt, ntc + q, lane);
                acc[q] = wmma_bf16(a, b, acc[q]);
            }
        }
        #pragma unroll
        for (int q = 0; q < 4; ++q) {
            int col = (ntc + q) * 16 + (lane & 15);
            #pragma unroll
            for (int i = 0; i < 8; ++i) {
                size_t off = (size_t)(m0 + rb + i) * HID + col;
                float v = fmaxf(binput[off] + acc[q][i], 0.f);
                msg_out[off] = f32_to_bf16_bits(v);
            }
        }
    }
}

// ---------------------------------------------------------------------------
// atom_hiddens = relu([fatoms, gather_sum(message, agraph)] @ W_o + b_o),
// atomicAdd into per-molecule sums. K padded 486 -> 512. Branchless epilogue.
// ---------------------------------------------------------------------------
__global__ __launch_bounds__(WAVES * 32)
void readout_kernel(const float* __restrict__ fatoms,
                    const unsigned short* __restrict__ msg,
                    const unsigned short* __restrict__ Wo_p,
                    const float* __restrict__ b_o,
                    const long long* __restrict__ agraph,
                    const long long* __restrict__ mol_idx,
                    float* __restrict__ out_sums,
                    int n_atoms, int n_tiles) {
    __shared__ unsigned short s_ain[WAVES][16][512];   // 131072 B
    int lane = threadIdx.x & 31;
    int wave = threadIdx.x >> 5;
    int tile = blockIdx.x * WAVES + wave;
    if (tile >= n_tiles) return;
    int a0 = tile * 16;

    for (int r = 0; r < 16; ++r) {
        int row = min(a0 + r, n_atoms - 1);
        for (int c = lane; c < ATOM_DIM; c += 32)
            s_ain[wave][r][c] = f32_to_bf16_bits(fatoms[(size_t)row * ATOM_DIM + c]);
        if (lane < 512 - (ATOM_DIM + HID))
            s_ain[wave][r][ATOM_DIM + HID + lane] = 0;
        const long long* ag = agraph + (size_t)row * MAX_NB;
        float a0f[7], a1f[7];
        #pragma unroll
        for (int c = 0; c < 7; ++c) { a0f[c] = 0.f; a1f[c] = 0.f; }
        #pragma unroll
        for (int k = 0; k < MAX_NB; ++k) {
            size_t nb = (size_t)ag[k];
            const unsigned* sp = (const unsigned*)(msg + nb * HID);
            #pragma unroll
            for (int c = 0; c < 7; ++c) {
                unsigned u = sp[c * 32 + lane];
                a0f[c] += bf16lo_f32(u);
                a1f[c] += bf16hi_f32(u);
            }
        }
        #pragma unroll
        for (int c = 0; c < 7; ++c) {
            // bf16 col = 38 + 2*(c*32+lane): even -> 4B-aligned u32 store
            *(unsigned*)&s_ain[wave][r][ATOM_DIM + 2 * (c * 32 + lane)] = pack_bf16x2(a0f[c], a1f[c]);
        }
    }
    asm volatile("s_wait_dscnt 0" ::: "memory");

    int rb = (lane < 16) ? 0 : 8;
    const unsigned short* arow_base = &s_ain[wave][lane & 15][(lane < 16) ? 0 : 8];
    #pragma unroll 1
    for (int ntc = 0; ntc < NTILES; ntc += 4) {
        v8f acc[4];
        #pragma unroll
        for (int q = 0; q < 4; ++q) acc[q] = (v8f){0.f,0.f,0.f,0.f,0.f,0.f,0.f,0.f};
        #pragma unroll 2
        for (int kt = 0; kt < 512 / 32; ++kt) {
            Frag a;
            a.q[0] = *(const uint4*)(arow_base + kt * 32);
            a.q[1] = *(const uint4*)(arow_base + kt * 32 + 16);
            #pragma unroll
            for (int q = 0; q < 4; ++q) {
                Frag b;
                load_bfrag(b, Wo_p, kt, ntc + q, lane);
                acc[q] = wmma_bf16(a, b, acc[q]);
            }
        }
        #pragma unroll
        for (int q = 0; q < 4; ++q) {
            int col = (ntc + q) * 16 + (lane & 15);
            float bias = b_o[col];
            #pragma unroll
            for (int i = 0; i < 8; ++i) {
                int row  = a0 + rb + i;
                int ok   = row < n_atoms;
                int crow = ok ? row : (n_atoms - 1);
                float v  = ok ? fmaxf(acc[q][i] + bias, 0.f) : 0.f;
                long long mol = mol_idx[crow];
                atomicAdd(&out_sums[(size_t)mol * HID + col], v);
            }
        }
    }
}

// ---------------------------------------------------------------------------
__global__ void zero_kernel(float* __restrict__ p, size_t n) {
    size_t i = (size_t)blockIdx.x * blockDim.x + threadIdx.x;
    if (i < n) p[i] = 0.f;
}
__global__ void counts_kernel(const long long* __restrict__ mol_idx,
                              float* __restrict__ counts, int n_atoms) {
    int i = blockIdx.x * blockDim.x + threadIdx.x;
    if (i < n_atoms) atomicAdd(&counts[mol_idx[i]], 1.0f);
}
__global__ void divide_kernel(float* __restrict__ out,
                              const float* __restrict__ counts, int n_total) {
    int i = blockIdx.x * blockDim.x + threadIdx.x;
    if (i < n_total) out[i] = out[i] / fmaxf(counts[i / HID], 1.0f);
}

// ---------------------------------------------------------------------------
extern "C" void kernel_launch(void* const* d_in, const int* in_sizes, int n_in,
                              void* d_out, int out_size, void* d_ws, size_t ws_size,
                              hipStream_t stream) {
    const float*     fatoms  = (const float*)d_in[0];
    const float*     fbonds  = (const float*)d_in[1];
    const float*     W_i     = (const float*)d_in[2];
    const float*     W_h     = (const float*)d_in[3];
    const float*     W_o     = (const float*)d_in[4];
    const float*     b_o     = (const float*)d_in[5];
    const long long* agraph  = (const long long*)d_in[6];
    const long long* bgraph  = (const long long*)d_in[7];
    const long long* mol_idx = (const long long*)d_in[8];

    int n_atoms = in_sizes[0] / ATOM_DIM;
    int n_bonds = in_sizes[1] / BOND_FDIM;
    int n_mols  = out_size / HID;
    float* out  = (float*)d_out;

    int btiles  = (n_bonds + 15) / 16;
    int atiles  = (n_atoms + 15) / 16;
    size_t nb_pad = (size_t)btiles * 16;   // row-padded bond count

    // ---- workspace carve (256B aligned) ----
    char* ws = (char*)d_ws;
    auto carve = [&](size_t bytes) -> void* {
        void* p = (void*)ws;
        ws += (bytes + 255) & ~(size_t)255;
        return p;
    };
    unsigned short* Wi_p   = (unsigned short*)carve((size_t)2  * NTILES * 512 * 2);
    unsigned short* Wh_p   = (unsigned short*)carve((size_t)14 * NTILES * 512 * 2);
    unsigned short* Wo_p   = (unsigned short*)carve((size_t)16 * NTILES * 512 * 2);
    float*          binput = (float*)carve(nb_pad * HID * 4);
    unsigned short* msgA   = (unsigned short*)carve(nb_pad * HID * 2);
    unsigned short* msgB   = (unsigned short*)carve(nb_pad * HID * 2);
    float*          counts = (float*)carve((size_t)n_mols * 4);

    // ---- pack weights into WMMA B-fragment order ----
    {
        int t;
        t = 2  * NTILES * 512; hipLaunchKernelGGL(pack_weight_kernel, dim3((t + 255) / 256), dim3(256), 0, stream, W_i, Wi_p, BOND_FDIM, 2);
        t = 14 * NTILES * 512; hipLaunchKernelGGL(pack_weight_kernel, dim3((t + 255) / 256), dim3(256), 0, stream, W_h, Wh_p, HID, 14);
        t = 16 * NTILES * 512; hipLaunchKernelGGL(pack_weight_kernel, dim3((t + 255) / 256), dim3(256), 0, stream, W_o, Wo_p, ATOM_DIM + HID, 16);
    }

    // ---- zero output sums and counts ----
    {
        size_t n = (size_t)n_mols * HID;
        hipLaunchKernelGGL(zero_kernel, dim3((unsigned)((n + 255) / 256)), dim3(256), 0, stream, out, n);
        hipLaunchKernelGGL(zero_kernel, dim3((unsigned)((n_mols + 255) / 256)), dim3(256), 0, stream, counts, (size_t)n_mols);
    }

    // ---- binput GEMM + message0 ----
    int bblocks = (btiles + WAVES - 1) / WAVES;
    hipLaunchKernelGGL(binput_kernel, dim3(bblocks), dim3(WAVES * 32), 0, stream,
                       fbonds, Wi_p, binput, msgA, n_bonds, btiles);

    // ---- 4 message-passing iterations (ping-pong) ----
    unsigned short* cur = msgA;
    unsigned short* nxt = msgB;
    for (int it = 0; it < N_ITERS; ++it) {
        hipLaunchKernelGGL(mpn_iter_kernel, dim3(bblocks), dim3(WAVES * 32), 0, stream,
                           cur, binput, Wh_p, bgraph, nxt, n_bonds, btiles);
        unsigned short* t = cur; cur = nxt; nxt = t;
    }
    // final message is in `cur`

    // ---- per-molecule counts ----
    hipLaunchKernelGGL(counts_kernel, dim3((n_atoms + 255) / 256), dim3(256), 0, stream,
                       mol_idx, counts, n_atoms);

    // ---- atom readout + segment sums ----
    int ablocks = (atiles + WAVES - 1) / WAVES;
    hipLaunchKernelGGL(readout_kernel, dim3(ablocks), dim3(WAVES * 32), 0, stream,
                       fatoms, cur, Wo_p, b_o, agraph, mol_idx, out, n_atoms, atiles);

    // ---- segment mean ----
    int ntot = n_mols * HID;
    hipLaunchKernelGGL(divide_kernel, dim3((ntot + 255) / 256), dim3(256), 0, stream,
                       out, counts, ntot);
}
